// Favor_24043226923302
// MI455X (gfx1250) — matrix operand order repaired and checked by reading, not compiled, not run on hardware
//
#include <hip/hip_runtime.h>
#include <math.h>

// ---------------------------------------------------------------------------
// FAVOR linear attention, fused for MI455X (gfx1250, wave32, WMMA).
//
// Fusion: p_q = (x@Wq+bq)@w = x@(Wq·w) + bq·w  (same for k), so q/k full
// projections are never materialized. Pass 1 computes, per 16-row t-tile:
//   [ v (64 cols) | p_q (8) | p_k (8) ]  =  x_tile @ Bfused   (5 WMMA n-tiles)
// then the cos/sin feature map, stores qP to workspace, and reduces
// S = kP^T v and kSum into global accumulators (LDS ds_add + global atomics).
// Pass 2: out = (qP@S) / (qP@kSum).   HBM-bound: x read once (64MB ~ 2.8us).
// ---------------------------------------------------------------------------

typedef __attribute__((ext_vector_type(16))) _Float16 v16h;
typedef __attribute__((ext_vector_type(8)))  float    v8f;

#define B_   4
#define T_   4096
#define E_   1024
#define DH_  64
#define M_   8
#define NT_  5              // 4 tiles of V cols + 1 tile [p_q | p_k]
#define KS_  (E_/32)        // 32 k-steps of 32
#define INV_SQRT_M 0.35355339059327373f

#define FRAG_N (NT_*KS_*32*16)   // 81920 f16 fragment elements
#define ZERO_N (B_*(16*DH_+16))  // 4160 floats: S accumulators + kSum
#define BIAS_N 80                // 64 (bv) + 16 (bq·w | bk·w)

// ---------------------------------------------------------------------------
// Prep: build fragment-ordered f16 B matrix [v | Wq·w | Wk·w], biases,
// and zero the S/kSum accumulators (must re-zero every launch).
// B-fragment layout (contiguous k, same as A in pass 1) (16-bit B 32x16): lane = N (0..15) x khalf,
// element j -> K = 16*(lane>=16) + j within the k-step.
// ---------------------------------------------------------------------------
__global__ void favor_prep(const float* __restrict__ Wq, const float* __restrict__ bq,
                           const float* __restrict__ Wk, const float* __restrict__ bk,
                           const float* __restrict__ Wv, const float* __restrict__ bv,
                           const float* __restrict__ w,
                           _Float16* __restrict__ Bfrag,
                           float* __restrict__ bias,
                           float* __restrict__ Szero) {
  int tid = blockIdx.x * 256 + threadIdx.x;
  if (tid < FRAG_N) {
    int j    = tid & 15;
    int lane = (tid >> 4) & 31;
    int ks   = (tid >> 9) & 31;
    int tile = tid >> 14;                      // 0..4
    int n    = lane & 15;
    int k    = ks * 32 + ((lane >> 4) << 4) + j;
    float val;
    if (tile < 4) {
      val = Wv[k * DH_ + tile * 16 + n];
    } else {                                   // fused p-projection: W{q,k}·w
      const float* Ws = (n < 8) ? Wq : Wk;
      int m = n & 7;
      float s = 0.f;
      for (int d = 0; d < DH_; ++d) s += Ws[k * DH_ + d] * w[d * M_ + m];
      val = s;
    }
    Bfrag[tid] = (_Float16)val;
  } else if (tid < FRAG_N + ZERO_N) {
    Szero[tid - FRAG_N] = 0.f;
  } else if (tid < FRAG_N + ZERO_N + BIAS_N) {
    int i = tid - FRAG_N - ZERO_N;
    float v;
    if (i < 64) {
      v = bv[i];
    } else {
      int m = i - 64;
      const float* bb = (m < 8) ? bq : bk;
      int mm = m & 7;
      float s = 0.f;
      for (int d = 0; d < DH_; ++d) s += bb[d] * w[d * M_ + mm];
      v = s;
    }
    bias[i] = v;
  }
}

// ---------------------------------------------------------------------------
// Pass 1: per wave, one 16-row t-tile. 5 WMMA accumulators over K=1024.
// ---------------------------------------------------------------------------
__global__ void __launch_bounds__(256)
favor_pass1(const float* __restrict__ x,
            const _Float16* __restrict__ Bfrag,
            const float* __restrict__ bias,
            float* __restrict__ qP,
            float* __restrict__ S_g,
            float* __restrict__ kSum_g) {
  __shared__ float kP_lds[8][16][17];   // per-wave kP tile (padded vs banks)
  __shared__ float S_lds[16 * DH_];
  __shared__ float kSum_lds[16];

  const int tid   = threadIdx.x;
  const int wave  = tid >> 5;
  const int lane  = tid & 31;
  const int b     = blockIdx.y;
  const int t0    = (blockIdx.x * 8 + wave) * 16;
  const int col   = lane & 15;
  const int khalf = lane >> 4;
  const int rbase = khalf * 8;

  for (int i = tid; i < 16 * DH_; i += 256) S_lds[i] = 0.f;
  if (tid < 16) kSum_lds[tid] = 0.f;

  v8f acc[NT_] = {};

  // A-matrix 16x32 f16 layout (contiguous k, as in Bfrag): lane holds row M=lane&15;
  // element j -> K = 16*khalf + j.
  const float* xrow = x + ((size_t)(b * T_ + t0 + col)) * E_;

  for (int ks = 0; ks < KS_; ++ks) {
    const float* xp = xrow + ks * 32 + (khalf << 4);
    v16h a;
#pragma unroll
    for (int j = 0; j < 8; ++j) a[j] = (_Float16)xp[j];
#pragma unroll
    for (int j = 0; j < 8; ++j) a[8 + j] = (_Float16)xp[8 + j];
    __builtin_prefetch(xp + 128, 0, 0);   // global_prefetch_b8, ~4 k-steps ahead

    const _Float16* bp = Bfrag + (size_t)ks * 512 + lane * 16;
#pragma unroll
    for (int t = 0; t < NT_; ++t) {
      v16h bf = *(const v16h*)(bp + (size_t)t * KS_ * 512);
      acc[t] = __builtin_amdgcn_wmma_f32_16x16x32_f16(
          false, a, false, bf, (short)0, acc[t], false, false);
    }
  }

  // Epilogue. C/D layout: VGPR r holds (M = rbase + r, N = col).
  float vreg[4][8];
#pragma unroll
  for (int t = 0; t < 4; ++t) {
    float bb = bias[t * 16 + col];
#pragma unroll
    for (int r = 0; r < 8; ++r) vreg[t][r] = acc[t][r] + bb;
  }
  {
    float bp = bias[64 + col];
#pragma unroll
    for (int r = 0; r < 8; ++r) {
      float p = acc[4][r] + bp;
      float c = __cosf(p) * INV_SQRT_M;
      float s = __sinf(p) * INV_SQRT_M;
      int row = rbase + r;
      if (col < 8) {                      // q features -> workspace
        float* q = qP + ((size_t)(b * T_ + t0 + row)) * 16;
        q[col]     = c;
        q[col + 8] = s;
      } else {                            // k features -> LDS for S reduction
        kP_lds[wave][row][col - 8] = c;
        kP_lds[wave][row][col]     = s;
      }
    }
  }
  __syncthreads();

  // kSum partial: lane handles feature f=col over its 8 rows.
  {
    float s = 0.f;
#pragma unroll
    for (int r = 0; r < 8; ++r) s += kP_lds[wave][rbase + r][col];
    atomicAdd(&kSum_lds[col], s);
  }
  // S partial: lane owns columns d = t*16+col of v, rows rbase..rbase+7.
#pragma unroll
  for (int t = 0; t < 4; ++t) {
    int d = t * 16 + col;
#pragma unroll
    for (int m = 0; m < 16; ++m) {
      float s = 0.f;
#pragma unroll
      for (int r = 0; r < 8; ++r) s += kP_lds[wave][rbase + r][m] * vreg[t][r];
      atomicAdd(&S_lds[m * DH_ + d], s);
    }
  }
  __syncthreads();

  for (int i = tid; i < 16 * DH_; i += 256)
    unsafeAtomicAdd(&S_g[b * 16 * DH_ + i], S_lds[i]);
  if (tid < 16)
    unsafeAtomicAdd(&kSum_g[b * 16 + tid], kSum_lds[tid]);
}

// ---------------------------------------------------------------------------
// Pass 2: out[b,t,d] = (qP[t,:]·S[:,d]) / (qP[t,:]·kSum). 4 rows x 64 d/block.
// ---------------------------------------------------------------------------
__global__ void __launch_bounds__(256)
favor_pass2(const float* __restrict__ qP, const float* __restrict__ S_g,
            const float* __restrict__ kSum_g, float* __restrict__ out) {
  __shared__ float S_s[16 * DH_];
  __shared__ float kS_s[16];
  __shared__ float qPs[4][16];
  int tid  = threadIdx.x;
  int b    = blockIdx.y;
  int row0 = blockIdx.x * 4;
  for (int i = tid; i < 16 * DH_; i += 256) S_s[i] = S_g[b * 16 * DH_ + i];
  if (tid < 16) kS_s[tid] = kSum_g[b * 16 + tid];
  if (tid < 64)
    qPs[tid >> 4][tid & 15] =
        qP[((size_t)(b * T_ + row0 + (tid >> 4))) * 16 + (tid & 15)];
  __syncthreads();
  int r = tid >> 6;
  int d = tid & 63;
  float D = 0.f, N = 0.f;
#pragma unroll
  for (int m = 0; m < 16; ++m) {
    float qv = qPs[r][m];
    D += qv * kS_s[m];
    N += qv * S_s[m * DH_ + d];
  }
  out[((size_t)(b * T_ + row0 + r)) * DH_ + d] = N / D;
}

// ---------------------------------------------------------------------------
extern "C" void kernel_launch(void* const* d_in, const int* in_sizes, int n_in,
                              void* d_out, int out_size, void* d_ws, size_t ws_size,
                              hipStream_t stream) {
  const float* x  = (const float*)d_in[0];
  const float* Wq = (const float*)d_in[1];
  const float* bq = (const float*)d_in[2];
  const float* Wk = (const float*)d_in[3];
  const float* bk = (const float*)d_in[4];
  const float* Wv = (const float*)d_in[5];
  const float* bv = (const float*)d_in[6];
  const float* w  = (const float*)d_in[7];
  float* out = (float*)d_out;

  // Workspace layout (~1.23 MB total, assumed <= ws_size):
  //   [0, 163840)           Bfrag (f16 fragments, 5 tiles x 32 ksteps)
  //   [163840, 180224)      S accumulators  4 x 16 x 64 f32
  //   [180224, 180480)      kSum            4 x 16 f32   (contiguous w/ S)
  //   [180480, 180800)      bias            80 f32
  //   [180800, +1 MB)       qP              4 x 4096 x 16 f32
  char* ws = (char*)d_ws;
  _Float16* Bfrag = (_Float16*)(ws);
  float* S_g  = (float*)(ws + 163840);
  float* kSum = (float*)(ws + 163840 + 16384);
  float* bias = (float*)(ws + 163840 + 16384 + 256);
  float* qP   = (float*)(ws + 163840 + 16384 + 256 + 320);

  int prep_threads = FRAG_N + ZERO_N + BIAS_N;        // 86160
  int prep_blocks  = (prep_threads + 255) / 256;      // 337
  favor_prep<<<prep_blocks, 256, 0, stream>>>(Wq, bq, Wk, bk, Wv, bv, w,
                                              Bfrag, bias, S_g);

  dim3 g1(T_ / (8 * 16), B_);   // 32 x 4 blocks, 8 waves each
  favor_pass1<<<g1, 256, 0, stream>>>(x, Bfrag, bias, qP, S_g, kSum);

  dim3 g2(T_ / 4, B_);          // 1024 x 4 blocks
  favor_pass2<<<g2, 256, 0, stream>>>(qP, S_g, kSum, out);
}
